// CausalSelfAttention_87840671138478
// MI455X (gfx1250) — matrix operand order repaired and checked by reading, not compiled, not run on hardware
//
#include <hip/hip_runtime.h>

typedef __attribute__((ext_vector_type(16))) _Float16 v16h;
typedef __attribute__((ext_vector_type(8)))  _Float16 v8h;
typedef __attribute__((ext_vector_type(8)))  float    v8f;

constexpr int NH    = 16;    // query heads
constexpr int NKV   = 4;     // kv heads
constexpr int HD    = 128;   // head dim
constexpr int SEQ   = 2048;  // T
constexpr int BATCH = 4;

#define WMMA_F16(a, b, c) \
  __builtin_amdgcn_wmma_f32_16x16x32_f16(false, (a), false, (b), (short)0, (c), false, false)

// 16 contiguous halves -> v16h (two 16B loads)
__device__ __forceinline__ v16h ld16_contig(const _Float16* p) {
  v8h lo = *(const v8h*)p;
  v8h hi = *(const v8h*)(p + 8);
  return __builtin_shufflevector(lo, hi, 0, 1, 2, 3, 4, 5, 6, 7,
                                 8, 9, 10, 11, 12, 13, 14, 15);
}
// two contiguous 8-half runs at p and p+16 -> v16h (A-fragment pattern)
__device__ __forceinline__ v16h ld16_gap16(const _Float16* p) {
  v8h lo = *(const v8h*)p;
  v8h hi = *(const v8h*)(p + 16);
  return __builtin_shufflevector(lo, hi, 0, 1, 2, 3, 4, 5, 6, 7,
                                 8, 9, 10, 11, 12, 13, 14, 15);
}

// Async copy 16B from global to a per-lane LDS address (ASYNCcnt-tracked DMA).
__device__ __forceinline__ void async_g2l_b128(unsigned lds_byte_addr,
                                               const _Float16* gaddr) {
  asm volatile("global_load_async_to_lds_b128 %0, %1, off"
               :: "v"(lds_byte_addr), "v"(gaddr) : "memory");
}
__device__ __forceinline__ void wait_async0() {
  asm volatile("s_wait_asynccnt 0" ::: "memory");
}

// ---------------------------------------------------------------------------
// WMMA GEMM, f16 inputs, f32 accumulate/output.
// C[MxN] = A[MxK] @ B[KxN]; M%128==0, N%64==0, K%32==0.
// Block = 256 threads = 8 waves; block tile 128x64, wave tile 16x64 (4 accums).
// A tile staged via GLOBAL_LOAD_ASYNC_TO_LDS_B128 (no VALU in the copy path);
// B tile staged transposed (Bp[n][k]) so fragments are contiguous per lane.
// ---------------------------------------------------------------------------
__global__ __launch_bounds__(256) void gemm_wmma_f16(const _Float16* __restrict__ A,
                                                     const _Float16* __restrict__ B,
                                                     float* __restrict__ C,
                                                     int M, int N, int K) {
  __shared__ _Float16 As[128][40];  // [m][k]  (pad row to 80B)
  __shared__ _Float16 Bp[64][40];   // [n][k]  transposed (pad row to 80B)

  const int tid  = threadIdx.x;
  const int lane = tid & 31;
  const int wave = tid >> 5;
  const int blockM = blockIdx.y * 128;
  const int blockN = blockIdx.x * 64;
  const int lm = lane & 15;
  const int lh = lane >> 4;

  const v8f vzero = {};
  v8f acc[4];
  #pragma unroll
  for (int t = 0; t < 4; ++t) acc[t] = vzero;

  // Per-thread staging coordinates
  const int ar = tid >> 1;              // A row 0..127
  const int ac = (tid & 1) * 16;        // A half-col 0 or 16
  const unsigned aLds = (unsigned)(uintptr_t)&As[ar][ac];
  const int br = tid >> 3;              // B k-row 0..31
  const int bc = (tid & 7) * 8;         // B n-col 0..56

  for (int k0 = 0; k0 < K; k0 += 32) {
    // ---- A tile (128x32 halves) via async DMA: 2 x 16B chunks / thread ----
    {
      const _Float16* agp = A + (size_t)(blockM + ar) * K + k0 + ac;
      async_g2l_b128(aLds,      agp);
      async_g2l_b128(aLds + 16, agp + 8);
    }
    // ---- B tile (32x64) transposed into Bp[n][k] ----
    {
      const _Float16* src = B + (size_t)(k0 + br) * N + blockN + bc;
      if (k0 + 32 < K) __builtin_prefetch(src + (size_t)32 * N, 0, 1);
      const v8h bv = *(const v8h*)src;
      #pragma unroll
      for (int i = 0; i < 8; ++i) Bp[bc + i][br] = bv[i];
    }
    wait_async0();
    __syncthreads();

    // A fragment: two ds_load_b128 (halves k = lh*16 + 0..15, contiguous)
    const v16h af = ld16_contig(&As[wave * 16 + lm][lh * 16]);
    #pragma unroll
    for (int t = 0; t < 4; ++t) {
      // B fragment: lane lm = col n, halves k = lh*16 + 0..15 (contiguous)
      const v16h bf = ld16_contig(&Bp[t * 16 + lm][lh * 16]);
      acc[t] = WMMA_F16(af, bf, acc[t]);
    }
    __syncthreads();
  }

  // C/D layout: element [m][n]: n = lm, m = r + 8*lh
  const int mbase = blockM + wave * 16 + lh * 8;
  #pragma unroll
  for (int t = 0; t < 4; ++t) {
    const int n0 = blockN + t * 16 + lm;
    #pragma unroll
    for (int r = 0; r < 8; ++r)
      C[(size_t)(mbase + r) * N + n0] = acc[t][r];
  }
}

// fp32 -> f16 elementwise (used once per operand; n % 256 == 0)
__global__ __launch_bounds__(256) void f32_to_f16(const float* __restrict__ in,
                                                  _Float16* __restrict__ out, int n) {
  const int i = blockIdx.x * blockDim.x + threadIdx.x;
  if (i < n) out[i] = (_Float16)in[i];
}

// ---------------------------------------------------------------------------
// Fused RoPE + RMSNorm. One wave per (token, head); heads 0..15 = q, 16..19 = k.
// Lane holds dims {lane, lane+32, lane+64, lane+96}: RoPE pairs (d, d+64)
// are lane-local. Wave32 shuffle reduction for mean-of-squares. Writes f16.
// ---------------------------------------------------------------------------
__global__ __launch_bounds__(128) void rope_rms(const float* __restrict__ qf,
                                                const float* __restrict__ kf,
                                                const float* __restrict__ cosb,
                                                const float* __restrict__ sinb,
                                                _Float16* __restrict__ q16,
                                                _Float16* __restrict__ k16) {
  const int gwave = (int)((blockIdx.x * blockDim.x + threadIdx.x) >> 5);
  const int lane  = threadIdx.x & 31;
  if (gwave >= BATCH * SEQ * (NH + NKV)) return;

  const int headIdx = gwave % (NH + NKV);
  const int tok     = gwave / (NH + NKV);   // b*SEQ + t
  const int t       = tok % SEQ;

  const float* src;
  _Float16* dst;
  if (headIdx < NH) {
    src = qf  + (size_t)tok * (NH * HD)  + headIdx * HD;
    dst = q16 + (size_t)tok * (NH * HD)  + headIdx * HD;
  } else {
    const int h = headIdx - NH;
    src = kf  + (size_t)tok * (NKV * HD) + h * HD;
    dst = k16 + (size_t)tok * (NKV * HD) + h * HD;
  }

  const float x0 = src[lane],      x1 = src[lane + 32];
  const float x2 = src[lane + 64], x3 = src[lane + 96];
  const float c0 = cosb[(size_t)t * 64 + lane];
  const float c1 = cosb[(size_t)t * 64 + lane + 32];
  const float s0 = sinb[(size_t)t * 64 + lane];
  const float s1 = sinb[(size_t)t * 64 + lane + 32];

  const float r0 = x0 * c0 - x2 * s0;
  const float r1 = x1 * c1 - x3 * s1;
  const float r2 = x0 * s0 + x2 * c0;
  const float r3 = x1 * s1 + x3 * c1;

  float ss = r0 * r0 + r1 * r1 + r2 * r2 + r3 * r3;
  #pragma unroll
  for (int m = 16; m >= 1; m >>= 1) ss += __shfl_xor(ss, m, 32);
  const float inv = rsqrtf(ss * (1.0f / 128.0f) + 1e-5f);

  dst[lane]      = (_Float16)(r0 * inv);
  dst[lane + 32] = (_Float16)(r1 * inv);
  dst[lane + 64] = (_Float16)(r2 * inv);
  dst[lane + 96] = (_Float16)(r3 * inv);
}

// V: fp32 (B*T, NKV*HD) -> f16 transposed (B*NKV*HD, SEQ) for contiguous
// per-lane B-fragment loads in the P@V WMMA.
__global__ __launch_bounds__(256) void v_transpose(const float* __restrict__ vf,
                                                   _Float16* __restrict__ vT,
                                                   int n) {
  const int i = blockIdx.x * blockDim.x + threadIdx.x;
  if (i >= n) return;
  const int tok = i >> 9;          // / (NKV*HD)
  const int c   = i & 511;
  const int b   = tok / SEQ, t = tok % SEQ;
  const int kvh = c >> 7,    d = c & 127;
  vT[(size_t)((b * NKV + kvh) * HD + d) * SEQ + t] = (_Float16)vf[i];
}

// ---------------------------------------------------------------------------
// Flash attention (causal, GQA 4:1). One wave per (b, h, 16-row q tile).
// kv chunks of 32: 8 WMMAs q.k^T, online softmax in f32, P bounced through
// LDS (C-frag -> A-frag), 8 WMMAs P@V^T. All fragments are 128-bit loads.
// Output written as f16 (feeds the f16 projection GEMM).
// ---------------------------------------------------------------------------
__global__ __launch_bounds__(128) void attn(const _Float16* __restrict__ q16,
                                            const _Float16* __restrict__ k16,
                                            const _Float16* __restrict__ vT,
                                            _Float16* __restrict__ y16) {
  __shared__ _Float16 P[4][16][40];   // per-wave 16x32 P tile (pad to 40)

  const int tid   = threadIdx.x;
  const int lane  = tid & 31;
  const int w     = tid >> 5;
  const int gwave = (int)((blockIdx.x * blockDim.x + tid) >> 5);
  const int qtile = gwave & 127;             // SEQ/16 tiles
  const int h     = (gwave >> 7) & 15;
  const int b     = gwave >> 11;
  const int kvh   = h >> 2;                  // n_rep = 4
  const int qb    = qtile * 16;
  const int lm    = lane & 15;
  const int lh    = lane >> 4;
  const float scale = 0.08838834764831845f;  // 1/sqrt(128)

  // q A-fragments: 4 chunks of K=32 over head dim 128 (vector loads)
  v16h qa[4];
  {
    const _Float16* qrow = q16 + (size_t)(b * SEQ + qb + lm) * (NH * HD) + h * HD;
    #pragma unroll
    for (int ch = 0; ch < 4; ++ch)
      qa[ch] = ld16_contig(qrow + ch * 32 + lh * 16);
  }

  const v8f vzero = {};
  v8f o[8];
  #pragma unroll
  for (int i = 0; i < 8; ++i) o[i] = vzero;
  float rm[8], rl[8];
  #pragma unroll
  for (int r = 0; r < 8; ++r) { rm[r] = -3.0e38f; rl[r] = 0.0f; }

  for (int c = 0; c < qb + 16; c += 32) {
    // ---- scores: S(16x32) = q . K^T, two 16x16 tiles ----
    v8f s0 = vzero, s1 = vzero;
    #pragma unroll
    for (int t = 0; t < 2; ++t) {
      const _Float16* krow =
          k16 + (size_t)(b * SEQ + c + t * 16 + lm) * (NKV * HD) + kvh * HD;
      #pragma unroll
      for (int ch = 0; ch < 4; ++ch) {
        const v16h bf = ld16_contig(krow + ch * 32 + lh * 16);
        if (t == 0) s0 = WMMA_F16(qa[ch], bf, s0);
        else        s1 = WMMA_F16(qa[ch], bf, s1);
      }
    }

    // ---- online softmax per row (rows live across 16 lanes) ----
    #pragma unroll
    for (int r = 0; r < 8; ++r) {
      const int qrow = qb + r + 8 * lh;
      float e0 = s0[r] * scale;
      float e1 = s1[r] * scale;
      if (qrow < c + lm)      e0 = -3.0e38f;
      if (qrow < c + 16 + lm) e1 = -3.0e38f;

      float mx = fmaxf(e0, e1);
      #pragma unroll
      for (int m = 8; m >= 1; m >>= 1) mx = fmaxf(mx, __shfl_xor(mx, m, 16));
      const float mnew = fmaxf(rm[r], mx);
      const float p0 = __expf(e0 - mnew);
      const float p1 = __expf(e1 - mnew);
      float rowsum = p0 + p1;
      #pragma unroll
      for (int m = 8; m >= 1; m >>= 1) rowsum += __shfl_xor(rowsum, m, 16);
      const float corr = __expf(rm[r] - mnew);
      rl[r] = rl[r] * corr + rowsum;
      rm[r] = mnew;
      #pragma unroll
      for (int dt = 0; dt < 8; ++dt) o[dt][r] *= corr;

      // stash P: P[m][n], m = r + 8*lh, n = t*16 + lm
      P[w][r + 8 * lh][lm]      = (_Float16)p0;
      P[w][r + 8 * lh][16 + lm] = (_Float16)p1;
    }
    asm volatile("s_wait_dscnt 0" ::: "memory");  // wave-local LDS write->read

    // ---- re-read P as A-fragment (two ds_load_b128) ----
    const v16h pf = ld16_contig(&P[w][lm][lh * 16]);

    // ---- O += P(16x32) @ V(32x128): V^T rows are contiguous in kv ----
    #pragma unroll
    for (int dt = 0; dt < 8; ++dt) {
      const _Float16* vrow =
          vT + (size_t)((b * NKV + kvh) * HD + dt * 16 + lm) * SEQ + c + lh * 16;
      const v16h bf = ld16_contig(vrow);
      o[dt] = WMMA_F16(pf, bf, o[dt]);
    }
    asm volatile("s_wait_dscnt 0" ::: "memory");  // order reads vs next writes
  }

  // ---- normalize and store y (f16) ----
  float invl[8];
  #pragma unroll
  for (int r = 0; r < 8; ++r) invl[r] = 1.0f / rl[r];
  #pragma unroll
  for (int dt = 0; dt < 8; ++dt) {
    #pragma unroll
    for (int r = 0; r < 8; ++r) {
      const int m = r + 8 * lh;
      y16[(size_t)(b * SEQ + qb + m) * (NH * HD) + h * HD + dt * 16 + lm] =
          (_Float16)(o[dt][r] * invl[r]);
    }
  }
}

// ---------------------------------------------------------------------------
extern "C" void kernel_launch(void* const* d_in, const int* in_sizes, int n_in,
                              void* d_out, int out_size, void* d_ws, size_t ws_size,
                              hipStream_t stream) {
  const float* x     = (const float*)d_in[0];
  const float* cosb  = (const float*)d_in[1];
  const float* sinb  = (const float*)d_in[2];
  const float* wq    = (const float*)d_in[3];
  const float* wk    = (const float*)d_in[4];
  const float* wv    = (const float*)d_in[5];
  const float* wproj = (const float*)d_in[6];
  float* out = (float*)d_out;

  const int M = BATCH * SEQ;   // 8192
  const int C = 2048;

  char* ws = (char*)d_ws;
  float*    qf   = (float*)(ws);                          // 64 MB
  _Float16* y16  = (_Float16*)(ws);                       // 32 MB (aliases qf; used after qf consumed)
  float*    kf   = (float*)(ws + (size_t)67108864);       // 16 MB
  float*    vf   = (float*)(ws + (size_t)83886080);       // 16 MB
  _Float16* q16  = (_Float16*)(ws + (size_t)100663296);   // 32 MB
  _Float16* k16  = (_Float16*)(ws + (size_t)134217728);   //  8 MB
  _Float16* vT   = (_Float16*)(ws + (size_t)142606336);   //  8 MB
  _Float16* x16  = (_Float16*)(ws + (size_t)150994944);   // 32 MB
  _Float16* wq16 = (_Float16*)(ws + (size_t)184549376);   //  8 MB
  _Float16* wk16 = (_Float16*)(ws + (size_t)192937984);   //  2 MB
  _Float16* wv16 = (_Float16*)(ws + (size_t)195035136);   //  2 MB
  _Float16* wp16 = (_Float16*)(ws + (size_t)197132288);   //  8 MB
  // total 205,520,896 bytes of workspace

  dim3 blk(256);
  // one-time f16 conversions of activations/weights
  f32_to_f16<<<dim3((M * C) / 256),     blk, 0, stream>>>(x,     x16,  M * C);
  f32_to_f16<<<dim3((C * 2048) / 256),  blk, 0, stream>>>(wq,    wq16, C * 2048);
  f32_to_f16<<<dim3((C * 512) / 256),   blk, 0, stream>>>(wk,    wk16, C * 512);
  f32_to_f16<<<dim3((C * 512) / 256),   blk, 0, stream>>>(wv,    wv16, C * 512);
  f32_to_f16<<<dim3((2048 * C) / 256),  blk, 0, stream>>>(wproj, wp16, 2048 * C);
  // q/k/v projections (f16 in, f32 out)
  gemm_wmma_f16<<<dim3(2048 / 64, M / 128), blk, 0, stream>>>(x16, wq16, qf, M, 2048, C);
  gemm_wmma_f16<<<dim3(512  / 64, M / 128), blk, 0, stream>>>(x16, wk16, kf, M, 512,  C);
  gemm_wmma_f16<<<dim3(512  / 64, M / 128), blk, 0, stream>>>(x16, wv16, vf, M, 512,  C);
  // RoPE + RMSNorm (q,k) -> f16 ; V -> f16 transposed
  rope_rms<<<dim3((M * (NH + NKV) * 32) / 128), dim3(128), 0, stream>>>(
      qf, kf, cosb, sinb, q16, k16);
  v_transpose<<<dim3((M * NKV * HD) / 256), blk, 0, stream>>>(vf, vT, M * NKV * HD);
  // flash attention -> y16 (reuses qf storage after qf is consumed)
  attn<<<dim3((M * NH / 16) * 32 / 128), dim3(128), 0, stream>>>(q16, k16, vT, y16);
  // output projection
  gemm_wmma_f16<<<dim3(2048 / 64, M / 128), blk, 0, stream>>>(y16, wp16, out, M, 2048, C);
}
